// Caterpillar_12592844112188
// MI455X (gfx1250) — matrix operand order repaired and checked by reading, not compiled, not run on hardware
//
#include <hip/hip_runtime.h>
#include <cstdint>

// ---------------- problem constants (from reference) ----------------
#define NN    2
#define NB_   4096
#define DM_   1024
#define HH    8
#define DK_   64
#define DV_   64
#define RR    4
#define LL    32
#define T0_   32
#define TT_   (T0_ + NB_)      // 4128
#define SS_   (NB_ / LL)       // 128 scan steps
#define M1    (NN * NB_)       // 8192 rows
#define CO1   1600             // padded fused-projection width (1568 used)
#define RECLEN (LL + NB_)      // 4128 recurrent states per (n,r)
#define RECPAD 4160            // recVT padded row length (shifted by -1, zero pad tail)

typedef __bf16 bf16;
typedef __attribute__((ext_vector_type(16))) __bf16 v16bf;
typedef __attribute__((ext_vector_type(8)))  __bf16 v8bf;
typedef __attribute__((ext_vector_type(8)))  float  v8f;

// WMMA 16x16x32 bf16 fragment maps (CDNA5 ISA 7.12.2, wave32):
// A: lane&15 = M row; elements 0-15 -> K = hi*16+0..15 (two 16B runs; code's order, not the ISA table)
// B: lane&15 = N col; elements 0-15 -> K = hi*16+0..15                           (one 32B run; code's order, same as A)
// C: element v -> M = v + hi*8 ; N = lane&15
static __device__ __forceinline__ v16bf cat8(v8bf a, v8bf b) {
  return __builtin_shufflevector(a, b, 0,1,2,3,4,5,6,7,8,9,10,11,12,13,14,15);
}
static __device__ __forceinline__ v8f wmma_bf16(v16bf a, v16bf b, v8f c) {
  return __builtin_amdgcn_wmma_f32_16x16x32_bf16(false, a, false, b, (short)0, c, false, false);
}

// CDNA5 async global->LDS copy (ISA 15.18.3 op 98, tracked by ASYNCcnt).
// LDS address = low 32 bits of the generic pointer (LDS aperture has zero low bits, ISA 10.2).
static __device__ __forceinline__ void async_b128(void* lds_ptr, const void* gptr) {
  unsigned lds_off = (unsigned)(uintptr_t)lds_ptr;
  unsigned long long ga = (unsigned long long)(uintptr_t)gptr;
  asm volatile("global_load_async_to_lds_b128 %0, %1, off" :: "v"(lds_off), "v"(ga) : "memory");
}
static __device__ __forceinline__ void wait_async0() {
  asm volatile("s_wait_asynccnt 0x0" ::: "memory");
}

// ---------------- kernel 1: convert X slice to bf16 (8-wide) ----------------
__global__ __launch_bounds__(256) void k_pack_x(const float* __restrict__ X, bf16* __restrict__ Xb) {
  int idx8 = (blockIdx.x * 256 + threadIdx.x) * 8;    // M1*DM/8 threads
  int c = idx8 & (DM_ - 1);
  int m = idx8 >> 10;
  int n = m >> 12, t = m & (NB_ - 1);
  v8f x = *(const v8f*)(X + (size_t)n * TT_ * DM_ + (size_t)(T0_ + t) * DM_ + c);
  v8bf o;
#pragma unroll
  for (int e = 0; e < 8; ++e) o[e] = (bf16)x[e];
  *(v8bf*)(Xb + idx8) = o;
}

// ---------------- kernel 2: fused K|V|Q|G weights, N-major (native layout, 8-wide) ----------
__global__ __launch_bounds__(256) void k_pack_w1(const float* __restrict__ wK, const float* __restrict__ wV,
                                                 const float* __restrict__ wQ, const float* __restrict__ wG,
                                                 bf16* __restrict__ W1T) {
  int idx8 = (blockIdx.x * 256 + threadIdx.x) * 8;    // CO1*DM/8 threads
  v8bf o;
  if (idx8 < 1568 * 1024) {
    const float* src;
    if (idx8 < 512 * 1024)        src = wK + idx8;
    else if (idx8 < 1024 * 1024)  src = wV + (idx8 - 512 * 1024);
    else if (idx8 < 1536 * 1024)  src = wQ + (idx8 - 1024 * 1024);
    else                          src = wG + (idx8 - 1536 * 1024);
    v8f x = *(const v8f*)src;
#pragma unroll
    for (int e = 0; e < 8; ++e) o[e] = (bf16)x[e];
  } else {
#pragma unroll
    for (int e = 0; e < 8; ++e) o[e] = (bf16)0.f;
  }
  *(v8bf*)(W1T + idx8) = o;
}

// ---------------- kernel 3: w_O transposed to N-major [1024][512] ----------------
__global__ __launch_bounds__(256) void k_pack_wo(const float* __restrict__ wO, bf16* __restrict__ WOT) {
  int idx8 = (blockIdx.x * 256 + threadIdx.x) * 8;    // 1024*512/8 threads
  int j  = idx8 >> 9;                                 // output column of GEMM-2 (DM)
  int c0 = idx8 & 511;                                // contraction dim (H*DV)
  v8bf o;
#pragma unroll
  for (int e = 0; e < 8; ++e) o[e] = (bf16)wO[(size_t)(c0 + e) * DM_ + j];
  *(v8bf*)(WOT + idx8) = o;
}

// ---------------- WMMA GEMM with async global->LDS double-buffered staging --------------
// C[M][Nd] = A[M][K](bf16,row-major) x BT[Nd][K](bf16,N-major)
// block = 8 waves; 64x64 block tile, k-stage 64; wave (wm,wn) owns a 16x32 output strip.
__global__ __launch_bounds__(256) void k_gemm_bf16(const bf16* __restrict__ A, const bf16* __restrict__ BT,
                                                   float* __restrict__ C, int M, int K, int Nd) {
  __shared__ __align__(16) bf16 As[2][64][72];        // padded rows: 144B stride
  __shared__ __align__(16) bf16 Bs[2][64][72];
  int tid  = threadIdx.x;
  int lane = tid & 31, w = tid >> 5;
  int lo = lane & 15, hi = lane >> 4;
  int wm = w & 3, wn = w >> 2;
  int row0b = blockIdx.y * 64;
  int col0b = blockIdx.x * 64;
  v8f acc0 = {0.f,0.f,0.f,0.f,0.f,0.f,0.f,0.f};
  v8f acc1 = acc0;

  auto stage = [&](int buf, int k0) {                 // copy 64x64 A and B tiles, 16B/lane/op
#pragma unroll
    for (int it = 0; it < 2; ++it) {
      int q   = tid + it * 256;                       // 512 chunks per tile
      int row = q >> 3, seg = (q & 7) * 8;
      async_b128(&As[buf][row][seg], A  + (size_t)(row0b + row) * K + k0 + seg);
      async_b128(&Bs[buf][row][seg], BT + (size_t)(col0b + row) * K + k0 + seg);
    }
  };

  int nk = K >> 6;
  stage(0, 0);
  for (int kt = 0; kt < nk; ++kt) {
    wait_async0();                                    // my copies for buffer kt&1 complete
    __syncthreads();                                  // everyone's copies visible
    if (kt + 1 < nk) stage((kt + 1) & 1, (kt + 1) << 6);   // overlap next stage with compute
    int b = kt & 1;
    const bf16* arow = &As[b][wm * 16 + lo][0];
    const bf16* b0r  = &Bs[b][wn * 32 + lo][0];
    const bf16* b1r  = &Bs[b][wn * 32 + 16 + lo][0];
#pragma unroll
    for (int kc = 0; kc < 64; kc += 32) {
      v16bf a   = cat8(*(const v8bf*)(arow + kc + hi * 16), *(const v8bf*)(arow + kc + hi * 16 + 8));
      v16bf bb0 = cat8(*(const v8bf*)(b0r + kc + hi * 16), *(const v8bf*)(b0r + kc + hi * 16 + 8));
      v16bf bb1 = cat8(*(const v8bf*)(b1r + kc + hi * 16), *(const v8bf*)(b1r + kc + hi * 16 + 8));
      acc0 = wmma_bf16(a, bb0, acc0);
      acc1 = wmma_bf16(a, bb1, acc1);
    }
  }
#pragma unroll
  for (int v = 0; v < 8; ++v) {
    int row = row0b + wm * 16 + v + hi * 8;
    C[(size_t)row * Nd + col0b + wn * 32 + lo]      = acc0[v];
    C[(size_t)row * Nd + col0b + wn * 32 + 16 + lo] = acc1[v];
  }
}

// ---------------- kernel 5: gate normalization ----------------
__global__ __launch_bounds__(256) void k_gnorm(const float* __restrict__ C1, const float* __restrict__ bG,
                                               float* __restrict__ Gn, float* __restrict__ Aarr) {
  int idx = blockIdx.x * 256 + threadIdx.x;           // N*R*NB threads
  int t = idx & (NB_ - 1);
  int r = (idx >> 12) & 3;
  int n = idx >> 14;
  size_t m = (size_t)(n * NB_ + t);
  float s[HH];
  float sum = 0.f;
#pragma unroll
  for (int h = 0; h < HH; ++h) {
    float x = C1[m * CO1 + 1536 + h * 4 + r] + bG[h * 4 + r];
    float sg = 1.f / (1.f + __expf(-x));
    s[h] = sg; sum += sg;
  }
  float denom = fmaxf(sum, 1.0f);
  size_t o = (size_t)((n * RR + r) * NB_ + t);
#pragma unroll
  for (int h = 0; h < HH; ++h) Gn[o * HH + h] = s[h] / denom;
  Aarr[o] = 1.f - sum / denom;
}

// ---------------- kernel 6: gated_K / gated_V (contract over heads) ----------------
__global__ __launch_bounds__(256) void k_gated(const float* __restrict__ C1, const float* __restrict__ Gn,
                                               float* __restrict__ gV, float* __restrict__ gK) {
  int idx = blockIdx.x * 256 + threadIdx.x;           // N*R*NB*64 threads
  int d = idx & 63;
  int t = (idx >> 6) & (NB_ - 1);
  int r = (idx >> 18) & 3;
  int n = idx >> 20;
  size_t o = (size_t)((n * RR + r) * NB_ + t);
  const float* gn = Gn + o * HH;
  const float* c1 = C1 + (size_t)(n * NB_ + t) * CO1;
  float av = 0.f, ak = 0.f;
#pragma unroll
  for (int h = 0; h < HH; ++h) {
    float g = gn[h];
    ak += g * c1[h * 64 + d];
    av += g * c1[512 + h * 64 + d];
  }
  gV[o * 64 + d] = av;
  gK[o * 64 + d] = ak;
}

// ---------------- kernel 7: linear recurrence over chunks ----------------
// recK: row-major [(n,r)][rec_idx][64]  (score B-fragments: head dim contiguous)
// recVT: transposed [(n,r)*64 + d][RECPAD], column i holds rec state idx i+1
__global__ __launch_bounds__(256) void k_scan(const float* __restrict__ Aarr,
                                              const float* __restrict__ gV, const float* __restrict__ gK,
                                              const float* __restrict__ initK, const float* __restrict__ initV,
                                              bf16* __restrict__ recK, bf16* __restrict__ recVT) {
  int tid = blockIdx.x * 256 + threadIdx.x;
  bool isK = tid >= 4096;
  int u = tid & 4095;
  int d0 = (u & 15) * 4;
  int l  = (u >> 4) & 31;
  int r  = (u >> 9) & 3;
  int n  = u >> 11;
  const float* init  = isK ? initK : initV;
  const float* gated = isK ? gK : gV;
  float st[4];
#pragma unroll
  for (int j = 0; j < 4; ++j) st[j] = init[(size_t)(r * LL + l) * 64 + d0 + j];
  size_t gbase = (size_t)(n * RR + r) * NB_;
  if (isK) {
    size_t recbase = (size_t)(n * RR + r) * RECLEN;
#pragma unroll
    for (int j = 0; j < 4; ++j) recK[(recbase + l) * 64 + d0 + j] = (bf16)st[j];  // init rows
    for (int s = 0; s < SS_; ++s) {
      int t = s * LL + l;
      float a = Aarr[gbase + t];
      const float* x = gated + (gbase + t) * 64 + d0;
#pragma unroll
      for (int j = 0; j < 4; ++j) st[j] = a * st[j] + x[j];
#pragma unroll
      for (int j = 0; j < 4; ++j) recK[(recbase + LL + t) * 64 + d0 + j] = (bf16)st[j];
    }
  } else {
    size_t rowb = (size_t)((n * RR + r) * 64 + d0) * RECPAD;
    if (l == 0) {                      // zero the pad tail once per (n,r,d-group)
#pragma unroll
      for (int j = 0; j < 4; ++j)
        for (int ii = RECLEN - 1; ii < RECPAD; ++ii)
          recVT[rowb + (size_t)j * RECPAD + ii] = (bf16)0.f;
    } else {
#pragma unroll
      for (int j = 0; j < 4; ++j) recVT[rowb + (size_t)j * RECPAD + (l - 1)] = (bf16)st[j];
    }
    for (int s = 0; s < SS_; ++s) {
      int t = s * LL + l;
      float a = Aarr[gbase + t];
      const float* x = gated + (gbase + t) * 64 + d0;
#pragma unroll
      for (int j = 0; j < 4; ++j) st[j] = a * st[j] + x[j];
#pragma unroll
      for (int j = 0; j < 4; ++j) recVT[rowb + (size_t)j * RECPAD + (LL + t - 1)] = (bf16)st[j];
    }
  }
}

// ---------------- kernel 8: sliding-window attention, 16 queries / block, 1 wave / head ------
__global__ __launch_bounds__(256) void k_attn(const float* __restrict__ C1,
                                              const bf16* __restrict__ recK, const bf16* __restrict__ recVT,
                                              bf16* __restrict__ Yb) {
  __shared__ __align__(16) bf16 Pb[HH][16 * 192];     // per-head probability tile
  int bid = blockIdx.x;
  int n   = bid >> 8;
  int tb0 = (bid & 255) << 4;
  int lane = threadIdx.x & 31;
  int h    = threadIdx.x >> 5;
  int lo = lane & 15, hi = lane >> 4;

  // Q fragments: 32B vector loads from f32 projections, convert to bf16
  const float* qb = C1 + (size_t)(n * NB_ + tb0 + lo) * CO1 + 1024 + h * 64;
  v8f q0 = *(const v8f*)(qb + hi * 16);
  v8f q1 = *(const v8f*)(qb + hi * 16 + 8);
  v8f q2 = *(const v8f*)(qb + 32 + hi * 16);
  v8f q3 = *(const v8f*)(qb + 32 + hi * 16 + 8);
  v16bf aQ0, aQ1;
#pragma unroll
  for (int e = 0; e < 8; ++e) {
    aQ0[e] = (bf16)q0[e]; aQ0[e + 8] = (bf16)q1[e];
    aQ1[e] = (bf16)q2[e]; aQ1[e + 8] = (bf16)q3[e];
  }

  // scores: per r, 3 key tiles of 16 (window span 47 rec rows, padded to 48)
  v8f acc[RR][3];
#pragma unroll
  for (int r = 0; r < RR; ++r)
#pragma unroll
    for (int nt = 0; nt < 3; ++nt) acc[r][nt] = (v8f){0.f,0.f,0.f,0.f,0.f,0.f,0.f,0.f};

#pragma unroll
  for (int r = 0; r < RR; ++r) {
    size_t rb = (size_t)(n * RR + r) * RECLEN;
#pragma unroll
    for (int nt = 0; nt < 3; ++nt) {
      int kk  = nt * 16 + lo;
      int idx = tb0 + 1 + kk; if (idx > RECLEN - 1) idx = RECLEN - 1;   // pad key, always masked
      const bf16* kb = recK + (rb + idx) * 64;
      v8bf t0 = *(const v8bf*)(kb + hi * 16);
      v8bf t1 = *(const v8bf*)(kb + hi * 16 + 8);
      v8bf t2 = *(const v8bf*)(kb + 32 + hi * 16);
      v8bf t3 = *(const v8bf*)(kb + 32 + hi * 16 + 8);
      v8f c = acc[r][nt];
      c = wmma_bf16(aQ0, cat8(t0, t1), c);
      c = wmma_bf16(aQ1, cat8(t2, t3), c);
      acc[r][nt] = c;
    }
  }

  // band mask + softmax (rows live in 16-lane halves)
  float rmax[8];
#pragma unroll
  for (int v = 0; v < 8; ++v) rmax[v] = -1e30f;
#pragma unroll
  for (int r = 0; r < RR; ++r)
#pragma unroll
    for (int nt = 0; nt < 3; ++nt)
#pragma unroll
      for (int v = 0; v < 8; ++v) {
        int i  = v + hi * 8;
        int kk = nt * 16 + lo;
        bool valid = (kk >= i) && (kk <= i + 31);
        float s = valid ? acc[r][nt][v] * 0.125f : -1e30f;   // 1/sqrt(DK)
        acc[r][nt][v] = s;
        rmax[v] = fmaxf(rmax[v], s);
      }
#pragma unroll
  for (int v = 0; v < 8; ++v)
#pragma unroll
    for (int msk = 1; msk < 16; msk <<= 1) rmax[v] = fmaxf(rmax[v], __shfl_xor(rmax[v], msk, 32));
  float rsum[8];
#pragma unroll
  for (int v = 0; v < 8; ++v) rsum[v] = 0.f;
#pragma unroll
  for (int r = 0; r < RR; ++r)
#pragma unroll
    for (int nt = 0; nt < 3; ++nt)
#pragma unroll
      for (int v = 0; v < 8; ++v) {
        float e = __expf(acc[r][nt][v] - rmax[v]);
        acc[r][nt][v] = e;
        rsum[v] += e;
      }
#pragma unroll
  for (int v = 0; v < 8; ++v)
#pragma unroll
    for (int msk = 1; msk < 16; msk <<= 1) rsum[v] += __shfl_xor(rsum[v], msk, 32);

  // normalized probs -> LDS (16 x 192, key = r*48 + nt*16 + col)
#pragma unroll
  for (int r = 0; r < RR; ++r)
#pragma unroll
    for (int nt = 0; nt < 3; ++nt)
#pragma unroll
      for (int v = 0; v < 8; ++v) {
        int row = v + hi * 8;
        Pb[h][row * 192 + r * 48 + nt * 16 + lo] = (bf16)(acc[r][nt][v] / rsum[v]);
      }

  // Y = P x win_V : 192-key contraction (6 chunks of 32), 4 output dim tiles
  v8f y[4];
#pragma unroll
  for (int nt = 0; nt < 4; ++nt) y[nt] = (v8f){0.f,0.f,0.f,0.f,0.f,0.f,0.f,0.f};
#pragma unroll
  for (int kg = 0; kg < 6; ++kg) {
    v8bf p0 = *(const v8bf*)&Pb[h][lo * 192 + kg * 32 + hi * 16];        // ds_load_b128
    v8bf p1 = *(const v8bf*)&Pb[h][lo * 192 + kg * 32 + hi * 16 + 8];
    v16bf ap = cat8(p0, p1);
    int s0 = kg * 32 + hi * 16;          // first 8-run of B K-indices
    int s1 = s0 + 8;                     // second 8-run (never straddles an r boundary)
    int r0 = s0 / 48, r1 = s1 / 48;
    int i0 = tb0 + s0 - 48 * r0;
    int i1 = tb0 + s1 - 48 * r1;
#pragma unroll
    for (int nt = 0; nt < 4; ++nt) {
      int d = nt * 16 + lo;
      const bf16* vr0 = recVT + (size_t)((n * RR + r0) * 64 + d) * RECPAD;
      const bf16* vr1 = recVT + (size_t)((n * RR + r1) * 64 + d) * RECPAD;
      v8bf w0 = *(const v8bf*)(vr0 + i0);
      v8bf w1 = *(const v8bf*)(vr1 + i1);
      y[nt] = wmma_bf16(ap, cat8(w0, w1), y[nt]);
    }
  }
#pragma unroll
  for (int nt = 0; nt < 4; ++nt)
#pragma unroll
    for (int v = 0; v < 8; ++v) {
      int row = v + hi * 8;
      Yb[(size_t)(n * NB_ + tb0 + row) * 512 + h * 64 + nt * 16 + lo] = (bf16)y[nt][v];
    }
}

// ---------------- launcher ----------------
extern "C" void kernel_launch(void* const* d_in, const int* in_sizes, int n_in,
                              void* d_out, int out_size, void* d_ws, size_t ws_size,
                              hipStream_t stream) {
  const float* X     = (const float*)d_in[0];
  const float* wG    = (const float*)d_in[1];
  const float* bG    = (const float*)d_in[2];
  const float* wK    = (const float*)d_in[3];
  const float* wV    = (const float*)d_in[4];
  const float* wQ    = (const float*)d_in[5];
  const float* wO    = (const float*)d_in[6];
  const float* initK = (const float*)d_in[7];
  const float* initV = (const float*)d_in[8];
  float* out = (float*)d_out;

  char* ws = (char*)d_ws;
  bf16*  Xb    = (bf16*)(ws + 0);            // 16.8 MB
  bf16*  W1T   = (bf16*)(ws + 16777216);     //  3.3 MB  (N-major fused weights)
  bf16*  WOT   = (bf16*)(ws + 20054016);     //  1.0 MB  (N-major w_O)
  float* C1    = (float*)(ws + 21102592);    // 52.4 MB  (fused K|V|Q|G projections)
  float* Gn    = (float*)(ws + 73531392);    //  1.0 MB
  float* Aarr  = (float*)(ws + 74579968);    //  0.13 MB
  float* gV    = (float*)(ws + 74711040);    //  8.4 MB
  float* gK    = (float*)(ws + 83099648);    //  8.4 MB
  bf16*  recK  = (bf16*)(ws + 91488256);     //  4.2 MB
  bf16*  recVT = (bf16*)(ws + 95715328);     //  4.3 MB (transposed, padded)
  bf16*  Yb    = (bf16*)(ws + 99975168);     //  8.4 MB -> total ~108 MB (fits 192MB L2)

  k_pack_x <<<(M1 * DM_) / (256 * 8), 256, 0, stream>>>(X, Xb);
  k_pack_w1<<<(CO1 * DM_) / (256 * 8), 256, 0, stream>>>(wK, wV, wQ, wG, W1T);
  k_pack_wo<<<(DM_ * 512) / (256 * 8), 256, 0, stream>>>(wO, WOT);

  // fused projection GEMM: 8192 x 1600 x 1024 (bf16 WMMA, async-LDS staged)
  k_gemm_bf16<<<dim3(CO1 / 64, M1 / 64), 256, 0, stream>>>(Xb, W1T, C1, M1, DM_, CO1);

  k_gnorm<<<(NN * RR * NB_) / 256, 256, 0, stream>>>(C1, bG, Gn, Aarr);
  k_gated<<<(NN * RR * NB_ * 64) / 256, 256, 0, stream>>>(C1, Gn, gV, gK);
  k_scan <<<8192 / 256, 256, 0, stream>>>(Aarr, gV, gK, initK, initV, recK, recVT);

  // windowed attention: one block per (n, 16-query tile), one wave per head
  k_attn<<<NN * (NB_ / 16), 256, 0, stream>>>(C1, recK, recVT, Yb);

  // output GEMM: 8192 x 1024 x 512 -> f32 d_out
  k_gemm_bf16<<<dim3(DM_ / 64, M1 / 64), 256, 0, stream>>>(Yb, WOT, out, M1, 512, DM_);
}
